// LowPassFilter_53661321396356
// MI455X (gfx1250) — matrix operand order repaired and checked by reading, not compiled, not run on hardware
//
#include <hip/hip_runtime.h>

typedef __attribute__((ext_vector_type(16))) _Float16 v16h;
typedef __attribute__((ext_vector_type(8)))  float    v8f;

#define LPF_N     32768
#define LPF_HALF  16
#define LPF_TAPS  33          // 2*HALF + 1
#define LPF_TILE  16          // outputs (t) per block, batches per wave
#define LPF_SEG   64          // 16 + 33 - 1 = 48, padded to 64 -> two K=32 WMMAs
#define SROW      65          // padded LDS row stride (floats) -> conflict-free gather
#define PI_F      3.14159265358979323846f

__global__ __launch_bounds__(128)
void lpf_wmma_kernel(const float* __restrict__ x,
                     const float* __restrict__ alphap,
                     const float* __restrict__ betap,
                     float* __restrict__ out,
                     int B, int n)
{
    __shared__ float    S[64 * SROW];                 // input segment: 64 batches x 64 window
    __shared__ float    fraw[LPF_TILE * LPF_TAPS];    // raw (unnormalized) filters
    __shared__ float    invs[LPF_TILE];               // 1 / row-sum
    __shared__ _Float16 filt16[LPF_TILE * LPF_TAPS];  // normalized + time-reversed, f16

    const int tid = threadIdx.x;
    const int t0  = blockIdx.x * LPF_TILE;
    const float alpha = *alphap;
    const float beta  = *betap;

    // ---- 1. synthesize the 16 per-sample windowed-sinc filters in LDS ----
    for (int idx = tid; idx < LPF_TILE * LPF_TAPS; idx += 128) {
        int j = idx / LPF_TAPS;
        int k = idx - j * LPF_TAPS;
        float t      = (float)(t0 + j);
        float cutoff = (PI_F * 0.25f + alpha * sinf(beta * t * (1.0f / 8000.0f)))
                       * (1.0f / (2.0f * PI_F));
        float wnd    = 0.5f - 0.5f * cosf(2.0f * PI_F * (float)k * (1.0f / 32.0f));
        float tm     = (float)(k - LPF_HALF);
        float xa     = 2.0f * PI_F * cutoff * tm;
        float snc    = (tm == 0.0f) ? 1.0f : (sinf(xa) / xa);
        fraw[idx]    = 2.0f * cutoff * wnd * snc;
    }
    __syncthreads();
    if (tid < LPF_TILE) {
        float s = 0.0f;
        for (int k = 0; k < LPF_TAPS; ++k) s += fraw[tid * LPF_TAPS + k];
        invs[tid] = 1.0f / s;
    }
    __syncthreads();
    for (int idx = tid; idx < LPF_TILE * LPF_TAPS; idx += 128) {
        int j = idx / LPF_TAPS;
        int k = idx - j * LPF_TAPS;
        // normalize, then reverse taps (reference does f[:, ::-1])
        filt16[idx] = (_Float16)(fraw[j * LPF_TAPS + (LPF_TAPS - 1 - k)] * invs[j]);
    }

    // ---- 2. stage the zero-padded input segment (coalesced 256B rows) ----
    const int nb = (B > 64) ? 64 : B;
    for (int idx = tid; idx < nb * LPF_SEG; idx += 128) {
        int b  = idx >> 6;
        int m  = idx & 63;
        int gt = t0 + m - LPF_HALF;
        float v = (gt >= 0 && gt < n) ? x[(size_t)b * n + gt] : 0.0f;
        S[b * SROW + m] = v;
    }
    __syncthreads();

    // ---- 3. WMMA: D(16b x 16t) = S(16b x 64m) * F(64m x 16t), banded F ----
    const int lane = tid & 31;
    const int h    = lane >> 4;    // half-wave
    const int r    = lane & 15;    // M for A / N (=j) for B, D

    // B fragments (F is shared by all batch tiles handled by this wave)
    // B layout: N = lane%16, K = e + 8h (e<8) / e + 8 + 8h (e>=8)
    v16h bf0, bf1;
    #pragma unroll
    for (int e = 0; e < 16; ++e) {
        int Kb = (e < 8) ? (e + 8 * h) : (e + 8 + 8 * h);
        int k0 = Kb - r;            // chunk0: m = Kb
        int k1 = Kb + 32 - r;       // chunk1: m = Kb + 32
        bf0[e] = (k0 >= 0 && k0 < LPF_TAPS) ? filt16[r * LPF_TAPS + k0] : (_Float16)0.0f;
        bf1[e] = (k1 >= 0 && k1 < LPF_TAPS) ? filt16[r * LPF_TAPS + k1] : (_Float16)0.0f;
    }

    const int wave = tid >> 5;
    for (int bt = wave; bt < (nb >> 4); bt += 4) {
        const int bb = bt * LPF_TILE;
        // A layout: M = lane%16, K = e + 8h (e<8) / e + 8 + 8h (e>=8)
        v16h af0, af1;
        #pragma unroll
        for (int e = 0; e < 16; ++e) {
            int Ka = (e < 8) ? (e + 8 * h) : (e + 8 + 8 * h);
            af0[e] = (_Float16)S[(bb + r) * SROW + Ka];
            af1[e] = (_Float16)S[(bb + r) * SROW + 32 + Ka];
        }
        v8f acc = {};
        acc = __builtin_amdgcn_wmma_f32_16x16x32_f16(false, af0, false, bf0,
                                                     (short)0, acc, false, false);
        acc = __builtin_amdgcn_wmma_f32_16x16x32_f16(false, af1, false, bf1,
                                                     (short)0, acc, false, false);
        // D layout: N = lane%16, element rr -> M = rr + 8h
        #pragma unroll
        for (int rr = 0; rr < 8; ++rr) {
            int M = rr + 8 * h;
            out[(size_t)(bb + M) * n + t0 + r] = acc[rr];
        }
    }
}

extern "C" void kernel_launch(void* const* d_in, const int* in_sizes, int n_in,
                              void* d_out, int out_size, void* d_ws, size_t ws_size,
                              hipStream_t stream) {
    const float* x     = (const float*)d_in[0];
    const float* alpha = (const float*)d_in[1];
    const float* beta  = (const float*)d_in[2];
    float* out = (float*)d_out;

    const int n = LPF_N;                 // reference fixes x to (64, 32768)
    const int B = in_sizes[0] / n;

    dim3 grid(n / LPF_TILE);             // 2048 blocks
    dim3 block(128);                     // 4 wave32 -> 4 batch tiles per block
    lpf_wmma_kernel<<<grid, block, 0, stream>>>(x, alpha, beta, out, B, n);
}
